// TA_HGAT_59055800320544
// MI455X (gfx1250) — matrix operand order repaired and checked by reading, not compiled, not run on hardware
//
#include <hip/hip_runtime.h>
#include <hip/hip_bf16.h>

#define N_TX    100000
#define N_EDGE  1000000
#define HID     64
#define HEADS   4

typedef __attribute__((ext_vector_type(16))) _Float16 v16h;
typedef __attribute__((ext_vector_type(8)))  float    v8f;
typedef __attribute__((ext_vector_type(2)))  _Float16 h2;

// ---------------------------------------------------------------------------
// Kernel 0: fuse weights.  Wc[32,256] = tx_W(32x64) @ lin_W(64x256)  (f16)
//           bc[256]       = tx_b(64)  @ lin_W(64x256)                (f32)
// ---------------------------------------------------------------------------
__global__ void k_fuse_weights(const float* __restrict__ tx_W,
                               const float* __restrict__ tx_b,
                               const float* __restrict__ lin_W,
                               _Float16* __restrict__ Wc,
                               float* __restrict__ bc) {
  int tid = blockIdx.x * 256 + threadIdx.x;
  if (tid < 32 * 256) {
    int k = tid >> 8, n = tid & 255;
    float acc = 0.f;
    #pragma unroll 8
    for (int j = 0; j < 64; ++j) acc += tx_W[k * 64 + j] * lin_W[j * 256 + n];
    Wc[k * 256 + n] = (_Float16)acc;
  } else if (tid < 32 * 256 + 256) {
    int n = tid - 32 * 256;
    float acc = 0.f;
    #pragma unroll 8
    for (int j = 0; j < 64; ++j) acc += tx_b[j] * lin_W[j * 256 + n];
    bc[n] = acc;
  }
}

// ---------------------------------------------------------------------------
// Kernel 1: x_h(100000,256) = f16( x_tx(100000,32) @ Wc(32,256) + bc )
// One wave -> one 16x16 tile, single v_wmma_f32_16x16x32_f16 (K=32 exact).
// Block = 8 waves covering 16 rows x 128 cols. Grid = (6250, 2).
// No divergence -> EXEC all ones as WMMA requires.
// ---------------------------------------------------------------------------
__global__ void k_gemm_wmma(const float* __restrict__ xtx,
                            const _Float16* __restrict__ Wc,
                            const float* __restrict__ bc,
                            _Float16* __restrict__ xh) {
  int wave = threadIdx.x >> 5;
  int lane = threadIdx.x & 31;
  int lh   = lane >> 4;        // half-wave: 0 or 1
  int ll   = lane & 15;
  int rowBase = blockIdx.x * 16;
  int colBase = blockIdx.y * 128 + wave * 16;
  int row = rowBase + ll;
  int col = colBase + ll;

  // A tile 16x32 f16: lane<16 -> M=ll, K in {0..7,16..23}; lane>=16 -> K+8
  v16h a;
  {
    const float* ap = xtx + row * 32 + lh * 8;
    #pragma unroll
    for (int e = 0; e < 8; ++e) a[e]     = (_Float16)ap[e];
    #pragma unroll
    for (int e = 0; e < 8; ++e) a[8 + e] = (_Float16)ap[16 + e];
  }
  // B tile 32x16 f16: lane<16 -> N=ll, K in {0..7,16..23}; lane>=16 -> K+8
  v16h b;
  {
    const _Float16* bp = Wc + (lh * 8) * 256 + col;
    #pragma unroll
    for (int e = 0; e < 16; ++e) b[e] = bp[((e & 7) + (e >> 3) * 16) * 256];
  }
  float bias = bc[col];
  v8f c;
  #pragma unroll
  for (int r = 0; r < 8; ++r) c[r] = bias;

  c = __builtin_amdgcn_wmma_f32_16x16x32_f16(
        /*neg_a=*/false, a, /*neg_b=*/false, b,
        /*c_mod=*/(short)0, c, /*reuse_a=*/false, /*reuse_b=*/false);

  #pragma unroll
  for (int r = 0; r < 8; ++r) {
    int m = rowBase + r + lh * 8;       // D: element r -> row r + 8*(lane>=16)
    xh[(long long)m * 256 + col] = (_Float16)c[r];
  }
}

// ---------------------------------------------------------------------------
// Kernel 2: per-node attention scores (read f16 x, accumulate f32)
//   s_i[n,h] = x[n,h,:]·att[h,0:64] ;  s_j[n,h] = x[n,h,:]·att[h,64:128]
// ---------------------------------------------------------------------------
__global__ void k_scores(const _Float16* __restrict__ xh,
                         const float* __restrict__ att,
                         float* __restrict__ si, float* __restrict__ sj) {
  int tid = blockIdx.x * blockDim.x + threadIdx.x;
  if (tid >= N_TX * HEADS) return;
  int h = tid & 3;
  const h2* xp = (const h2*)(xh + (long long)(tid >> 2) * 256 + h * 64);
  const float* ai = att + h * 128;
  const float* aj = ai + 64;
  float a = 0.f, b = 0.f;
  #pragma unroll 8
  for (int c = 0; c < 32; ++c) {
    h2 v = xp[c];
    float v0 = (float)v.x, v1 = (float)v.y;
    a += v0 * ai[2 * c] + v1 * ai[2 * c + 1];
    b += v0 * aj[2 * c] + v1 * aj[2 * c + 1];
  }
  si[tid] = a;
  sj[tid] = b;
}

// ---------------------------------------------------------------------------
// Kernel 3: zero agg (f16, must re-zero every launch; atomics accumulate)
// ---------------------------------------------------------------------------
__global__ void k_zero(float4* __restrict__ p, int n4) {
  int t = blockIdx.x * blockDim.x + threadIdx.x;
  if (t < n4) p[t] = make_float4(0.f, 0.f, 0.f, 0.f);
}

// ---------------------------------------------------------------------------
// Kernel 4: edge scatter. One wave per edge; each lane owns a channel PAIR.
//   alpha[h] = sigmoid(leaky_relu(s_i[dst,h]+s_j[src,h], 0.2) * exp(-beta*t))
//   agg[dst, 2l:2l+2] += sum_h alpha[h] * x[src, h, 2l:2l+2]
// One GLOBAL_ATOMIC_PK_ADD_F16 per lane per edge (32M packed atomics total,
// all hitting an L2-resident 12.8 MB accumulator).
// ---------------------------------------------------------------------------
__global__ void k_edges(const int* __restrict__ ei,
                        const float* __restrict__ et,
                        const _Float16* __restrict__ xh,
                        const float* __restrict__ si,
                        const float* __restrict__ sj,
                        const float* __restrict__ tbeta,
                        _Float16* __restrict__ agg) {
  int wave = threadIdx.x >> 5, lane = threadIdx.x & 31;
  long long e = (long long)blockIdx.x * 8 + wave;
  if (e >= N_EDGE) return;
  int src = ei[e];
  int dst = ei[N_EDGE + e];
  float beta = log1pf(expf(tbeta[0]));       // softplus
  float tw   = expf(-beta * et[e]);

  const h2* xs = (const h2*)(xh + (long long)src * 256);   // 128 pairs/node
  float acc0 = 0.f, acc1 = 0.f;
  #pragma unroll
  for (int h = 0; h < 4; ++h) {
    float al = si[dst * 4 + h] + sj[src * 4 + h];
    al = (al > 0.f) ? al : 0.2f * al;                 // leaky_relu 0.2
    float ah = 1.f / (1.f + expf(-al * tw));          // sigmoid
    h2 v = xs[h * 32 + lane];                         // channels 2l, 2l+1
    acc0 += ah * (float)v.x;
    acc1 += ah * (float)v.y;
  }
  h2 pk;
  pk.x = (_Float16)acc0;
  pk.y = (_Float16)acc1;
  unsigned pk_bits = __builtin_bit_cast(unsigned, pk);
  unsigned long long addr =
      (unsigned long long)(agg + (long long)dst * 64 + 2 * lane);
  // CDNA5 packed f16 atomic add, no return (STOREcnt-tracked)
  asm volatile("global_atomic_pk_add_f16 %0, %1, off"
               :: "v"(addr), "v"(pk_bits) : "memory");
}

// ---------------------------------------------------------------------------
// Kernel 5: out[n,:] = elu(agg[n,:]/4) @ cls_W + cls_b.  One wave per node.
// ---------------------------------------------------------------------------
__global__ void k_final(const _Float16* __restrict__ agg,
                        const float* __restrict__ cls_W,
                        const float* __restrict__ cls_b,
                        float* __restrict__ out) {
  int wave = threadIdx.x >> 5, lane = threadIdx.x & 31;
  int n = blockIdx.x * 8 + wave;
  if (n >= N_TX) return;
  float v0 = (float)agg[(long long)n * 64 + lane]      * 0.25f;
  float v1 = (float)agg[(long long)n * 64 + 32 + lane] * 0.25f;
  v0 = (v0 > 0.f) ? v0 : expm1f(v0);            // elu
  v1 = (v1 > 0.f) ? v1 : expm1f(v1);
  float p0 = v0 * cls_W[lane * 2]     + v1 * cls_W[(lane + 32) * 2];
  float p1 = v0 * cls_W[lane * 2 + 1] + v1 * cls_W[(lane + 32) * 2 + 1];
  #pragma unroll
  for (int off = 16; off; off >>= 1) {
    p0 += __shfl_xor(p0, off, 32);
    p1 += __shfl_xor(p1, off, 32);
  }
  if (lane == 0) {
    out[n * 2]     = p0 + cls_b[0];
    out[n * 2 + 1] = p1 + cls_b[1];
  }
}

// ---------------------------------------------------------------------------
extern "C" void kernel_launch(void* const* d_in, const int* in_sizes, int n_in,
                              void* d_out, int out_size, void* d_ws, size_t ws_size,
                              hipStream_t stream) {
  (void)in_sizes; (void)n_in; (void)out_size; (void)ws_size;
  // input order: x_user, x_tx, edge_index, edge_time, user_W, user_b,
  //              tx_W, tx_b, lin_W, att, time_beta, cls_W, cls_b
  const float* x_tx      = (const float*)d_in[1];
  const int*   edge_idx  = (const int*)  d_in[2];
  const float* edge_time = (const float*)d_in[3];
  const float* tx_W      = (const float*)d_in[6];
  const float* tx_b      = (const float*)d_in[7];
  const float* lin_W     = (const float*)d_in[8];
  const float* att       = (const float*)d_in[9];
  const float* time_beta = (const float*)d_in[10];
  const float* cls_W     = (const float*)d_in[11];
  const float* cls_b     = (const float*)d_in[12];
  float* out = (float*)d_out;

  // workspace layout (256B aligned), total ~67 MB -> L2-resident hot set
  char* ws = (char*)d_ws;
  _Float16* Wc  = (_Float16*)(ws);                                  //  16,384 B
  float*    bc  = (float*)(ws + 16384);                             //   1,024 B
  _Float16* xh  = (_Float16*)(ws + 17408);                          //  51,200,000 B
  float*    si  = (float*)(ws + 17408 + 51200000ULL);               //   1,600,000 B
  float*    sj  = si + (size_t)N_TX * HEADS;                        //   1,600,000 B
  _Float16* agg = (_Float16*)(ws + 17408 + 51200000ULL + 3200000ULL); // 12,800,000 B

  // 1) fuse tx_W@lin_W and tx_b@lin_W
  k_fuse_weights<<<33, 256, 0, stream>>>(tx_W, tx_b, lin_W, Wc, bc);
  // 2) x = x_tx @ Wc + bc   (WMMA, K=32 in one v_wmma_f32_16x16x32_f16)
  k_gemm_wmma<<<dim3(N_TX / 16, 2), 256, 0, stream>>>(x_tx, Wc, bc, xh);
  // 3) per-node attention scores
  k_scores<<<(N_TX * HEADS + 255) / 256, 256, 0, stream>>>(xh, att, si, sj);
  // 4) zero accumulator (12.8 MB = 800k float4)
  k_zero<<<(N_TX * HID * 2 / 16 + 255) / 256, 256, 0, stream>>>(
      (float4*)agg, N_TX * HID * 2 / 16);
  // 5) edge scatter with packed f16 global atomics (L2 resident)
  k_edges<<<N_EDGE / 8, 256, 0, stream>>>(edge_idx, edge_time, xh, si, sj,
                                          time_beta, agg);
  // 6) elu + classifier
  k_final<<<(N_TX + 7) / 8, 256, 0, stream>>>(agg, cls_W, cls_b, out);
}
